// ModelP_29059748725594
// MI455X (gfx1250) — matrix operand order repaired and checked by reading, not compiled, not run on hardware
//
#include <hip/hip_runtime.h>
#include <hip/hip_bf16.h>

// ---------------- model dims ----------------
static constexpr int NN   = 100000;           // vertices
static constexpr int MM   = 25000;            // hyperedges
static constexpr int KK   = 16;               // verts per edge
static constexpr int NPAD = 100096;           // 782 * 128  (rows padded for WMMA tiling)

typedef __attribute__((ext_vector_type(8)))  __bf16 v8bf;
typedef __attribute__((ext_vector_type(16))) __bf16 v16bf;
typedef __attribute__((ext_vector_type(8)))  float  v8f;

// ---------------- utility kernels ----------------
__global__ void k_zero(float* p, size_t n) {
  size_t t = (size_t)blockIdx.x * blockDim.x + threadIdx.x;
  if (t < n) p[t] = 0.0f;
}

__global__ void k_cvt(const float* __restrict__ s, __bf16* __restrict__ d, size_t n) {
  size_t t = (size_t)blockIdx.x * blockDim.x + threadIdx.x;
  if (t < n) d[t] = (__bf16)s[t];
}

// f32 [NN,C] -> bf16 [NPAD,C], pad rows zeroed (so WMMA tiles need no guards)
__global__ void k_cvtpad(const float* __restrict__ s, __bf16* __restrict__ d, int C) {
  size_t t = (size_t)blockIdx.x * blockDim.x + threadIdx.x;
  size_t tot = (size_t)NPAD * C;
  if (t >= tot) return;
  size_t row = t / (size_t)C;
  d[t] = (row < (size_t)NN) ? (__bf16)s[t] : (__bf16)0.0f;
}

// zero only the pad rows of a bf16 [NPAD,C] buffer (rows NN..NPAD)
__global__ void k_padzero(__bf16* d, int C) {
  size_t t = (size_t)blockIdx.x * blockDim.x + threadIdx.x;
  size_t tot = (size_t)(NPAD - NN) * C;
  if (t < tot) d[(size_t)NN * C + t] = (__bf16)0.0f;
}

__global__ void k_deg(const int* __restrict__ vidx, float* __restrict__ deg, int mk) {
  int t = blockIdx.x * blockDim.x + threadIdx.x;
  if (t < mk) atomicAdd(&deg[vidx[t]], 1.0f);
}

// ---------------- WMMA GEMM: C[N,O] = A[N,I] @ W[O,I]^T + bias (+C if acc) ----------------
// A bf16 [NPAD,I] row-major, W bf16 [O,I] row-major, C f32 [NPAD,O].
// Wave tile = 16 rows x 64 cols (4 accumulators; A fragment reused 4x per k-step).
// block = 256 (8 waves stacked along M, sharing the W stream); grid = (O/64, NPAD/128).
__global__ void k_gemm_wmma(const __bf16* __restrict__ A, const __bf16* __restrict__ W,
                            const float* __restrict__ bias, float* __restrict__ C,
                            int I, int O, int acc) {
  const int wave = threadIdx.x >> 5;
  const int lane = threadIdx.x & 31;
  const int g    = lane >> 4;       // half-group (0: lanes 0-15, 1: lanes 16-31)
  const int ln   = lane & 15;
  const size_t r0 = ((size_t)blockIdx.y * 8 + wave) * 16;
  const int    c0 = blockIdx.x * 64;

  v8f ac0 = {}, ac1 = {}, ac2 = {}, ac3 = {};
  union { v16bf v; v8bf h[2]; } a;

  for (int k0 = 0; k0 < I; k0 += 32) {
    // A fragment 16x32: lane (g, m=ln) holds K = g*16..g*16+15 contiguous
    const __bf16* ap = A + (r0 + ln) * (size_t)I + k0 + g * 16;
    a.h[0] = *(const v8bf*)(ap);
    a.h[1] = *(const v8bf*)(ap + 8);
    // B fragments 32x16 (B = W^T): lane holds col c0+j*16+ln, K = g*16..g*16+15 contiguous
    const __bf16* wp = W + (size_t)(c0 + ln) * I + k0 + g * 16;
    const size_t wstep = (size_t)16 * I;
    v16bf b0 = *(const v16bf*)(wp);
    v16bf b1 = *(const v16bf*)(wp + wstep);
    v16bf b2 = *(const v16bf*)(wp + 2 * wstep);
    v16bf b3 = *(const v16bf*)(wp + 3 * wstep);
    ac0 = __builtin_amdgcn_wmma_f32_16x16x32_bf16(false, a.v, false, b0, (short)0, ac0, false, false);
    ac1 = __builtin_amdgcn_wmma_f32_16x16x32_bf16(false, a.v, false, b1, (short)0, ac1, false, false);
    ac2 = __builtin_amdgcn_wmma_f32_16x16x32_bf16(false, a.v, false, b2, (short)0, ac2, false, false);
    ac3 = __builtin_amdgcn_wmma_f32_16x16x32_bf16(false, a.v, false, b3, (short)0, ac3, false, false);
  }

  v8f* accs[4] = { &ac0, &ac1, &ac2, &ac3 };
#pragma unroll
  for (int q = 0; q < 4; ++q) {
    const int col = c0 + q * 16 + ln;
    const float bv = bias ? bias[col] : 0.0f;
#pragma unroll
    for (int j = 0; j < 8; ++j) {
      size_t row = r0 + (size_t)j + (size_t)g * 8;  // D: VGPR j -> M=j (lanes 0-15), M=j+8 (16-31)
      size_t off = row * (size_t)O + col;
      float v = (*accs[q])[j] + bv;
      if (acc) v += C[off];
      C[off] = v;
    }
  }
}

// scalar GEMM for O=2 discriminator head (f32 in, f32 out)
__global__ void k_gemm_small(const float* __restrict__ A, const float* __restrict__ W,
                             const float* __restrict__ b, float* __restrict__ C, int I) {
  int t = blockIdx.x * blockDim.x + threadIdx.x;
  if (t >= NN * 2) return;
  int r = t >> 1, o = t & 1;
  const float* a = A + (size_t)r * I;
  const float* w = W + (size_t)o * I;
  float s = b[o];
  for (int i = 0; i < I; ++i) s += a[i] * w[i];
  C[t] = s;
}

// ---------------- BatchNorm batch stats (sum / sumsq per channel) ----------------
__global__ void k_colstats(const float* __restrict__ x, float* __restrict__ stat, int C) {
  int c = blockIdx.x * blockDim.x + threadIdx.x;
  if (c >= C) return;
  int r0 = blockIdx.y * 512;
  int r1 = r0 + 512; if (r1 > NN) r1 = NN;
  float s1 = 0.f, s2 = 0.f;
  for (int r = r0; r < r1; ++r) {
    float v = x[(size_t)r * C + c];
    s1 += v; s2 += v * v;
  }
  atomicAdd(&stat[c], s1);
  atomicAdd(&stat[C + c], s2);
}

// ---------------- hypergraph v2v with fused BatchNorm ----------------
// bn(h) = sc*h + sh is affine per channel, so sum_k bn(h_k) = sc * sum_k h_k + K*sh.
__global__ void k_v2v(const float* __restrict__ x, const int* __restrict__ vidx,
                      const float* __restrict__ we, float* __restrict__ accb,
                      const float* __restrict__ stat, const float* __restrict__ gam,
                      const float* __restrict__ bet, int C) {
  __shared__ int idx[KK];
  int m = blockIdx.y;
  if (threadIdx.x < KK) idx[threadIdx.x] = vidx[m * KK + threadIdx.x];
  __syncthreads();
  int c = blockIdx.x * blockDim.x + threadIdx.x;
  if (c >= C) return;
  float mean = stat[c] * (1.0f / NN);
  float var  = stat[C + c] * (1.0f / NN) - mean * mean;
  float sc   = gam[c] * rsqrtf(var + 1e-5f);
  float sh   = bet[c] - mean * sc;
  float s = 0.f;
#pragma unroll
  for (int k = 0; k < KK; ++k) s += x[(size_t)idx[k] * C + c];
  float wm = we[m];
  s = (sc * s) * (wm * (1.0f / KK)) + sh * wm;   // == bn-sum * w_e / K
#pragma unroll
  for (int k = 0; k < KK; ++k) atomicAdd(&accb[(size_t)idx[k] * C + c], s);
}

// finalize: divide by degree, optional relu, write f32 and/or padded-bf16 outputs
__global__ void k_vfin(const float* __restrict__ accb, const float* __restrict__ deg,
                       float* __restrict__ dstF, __bf16* __restrict__ dstB, int C, int relu) {
  size_t t = (size_t)blockIdx.x * blockDim.x + threadIdx.x;
  if (t >= (size_t)NN * C) return;
  int n = (int)(t / (size_t)C);
  float v = accb[t] / fmaxf(deg[n], 1.0f);
  if (relu) v = fmaxf(v, 0.0f);
  if (dstF) dstF[t] = v;
  if (dstB) dstB[t] = (__bf16)v;
}

// ---------------- elementwise ----------------
__global__ void k_sigmoid(float* a, size_t n) {
  size_t t = (size_t)blockIdx.x * blockDim.x + threadIdx.x;
  if (t < n) a[t] = 1.0f / (1.0f + __expf(-a[t]));
}
__global__ void k_tanh(float* a, size_t n) {
  size_t t = (size_t)blockIdx.x * blockDim.x + threadIdx.x;
  if (t < n) a[t] = tanhf(a[t]);
}
__global__ void k_mul(float* d, const float* a, const float* b, size_t n) {
  size_t t = (size_t)blockIdx.x * blockDim.x + threadIdx.x;
  if (t < n) d[t] = a[t] * b[t];
}
__global__ void k_outs(float* o, const float* z, const float* tt, const float* xe, size_t n) {
  size_t t = (size_t)blockIdx.x * blockDim.x + threadIdx.x;
  if (t < n) o[t] = z[t] * tt[t] + (1.0f - z[t]) * xe[t];
}

// attention-pooled prototype: 128 lanes (one per channel) x 128 rows per block
__global__ void k_proto(const float* __restrict__ outs, const float* __restrict__ stat,
                        float* __restrict__ proto) {
  int c = threadIdx.x;                       // 0..127
  float mean = stat[c] * (1.0f / NN);
  float vv = (stat[128 + c] - (float)NN * mean * mean) * (1.0f / (NN - 1));
  float inv = 1.0f / (4.0f * (vv + 0.001f));
  int r0 = blockIdx.x * 128;
  int r1 = r0 + 128; if (r1 > NN) r1 = NN;
  float s = 0.f;
  for (int r = r0; r < r1; ++r) {
    float v = outs[(size_t)r * 128 + c];
    float dd = (v - mean) * (v - mean);
    float ee = dd * inv + 0.5f;
    s += v * (1.0f / (1.0f + __expf(-ee)));
  }
  atomicAdd(&proto[c], s);
}

// ---------------- host orchestration ----------------
extern "C" void kernel_launch(void* const* d_in, const int* in_sizes, int n_in,
                              void* d_out, int out_size, void* d_ws, size_t ws_size,
                              hipStream_t stream) {
  // inputs (setup_inputs order)
  const float* X      = (const float*)d_in[0];
  const int*   vidx   = (const int*)  d_in[1];
  const float* we     = (const float*)d_in[2];
  const float* Wp[13]; const float* Bp[13];
  for (int i = 0; i < 13; ++i) { Wp[i] = (const float*)d_in[3 + 2*i]; Bp[i] = (const float*)d_in[4 + 2*i]; }
  // 0 enc1,1 enc2,2 fc,3 f1u,4 f2u,5 f1r,6 f2r,7 f1,8 f2,9 dis1,10 dis2,11 dec1,12 dec2
  const float* G[6]; const float* BT[6];
  for (int i = 0; i < 6; ++i) { G[i] = (const float*)d_in[29 + 2*i]; BT[i] = (const float*)d_in[30 + 2*i]; }
  // 0 enc1, 1 enc2, 2 dis1, 3 dis2, 4 dec1, 5 dec2

  float* out  = (float*)d_out;
  float* outs = out;                                  // [NN,128]
  float* xde  = out + (size_t)NN * 128;               // [NN,128]
  float* prot = out + (size_t)2 * NN * 128;           // [128]
  float* xdis = prot + 128;                           // [NN,2]

  // workspace bump allocator
  char* w = (char*)d_ws;
  auto alloc = [&](size_t bytes) { char* p = w; w += (bytes + 255) & ~(size_t)255; return p; };
  float*  deg  = (float*) alloc((size_t)NN * 4);
  float*  stat = (float*) alloc(4096);
  __bf16* abuf = (__bf16*)alloc((size_t)NPAD * 128 * 2);   // generic 128-ch bf16 GEMM input
  __bf16* x1b  = (__bf16*)alloc((size_t)NPAD * 512 * 2);   // 512-ch bf16 GEMM input
  float*  hbuf = (float*) alloc((size_t)NPAD * 512 * 4);
  float*  vbuf = (float*) alloc((size_t)NN * 512 * 4);
  float*  x1   = (float*) alloc((size_t)NN * 512 * 4);     // f32 hidden (dis2 scalar gemm input)
  float*  xe   = (float*) alloc((size_t)NN * 128 * 4);
  float*  xe1  = (float*) alloc((size_t)NPAD * 128 * 4);
  float*  t1   = (float*) alloc((size_t)NPAD * 128 * 4);
  float*  t2   = (float*) alloc((size_t)NPAD * 128 * 4);
  float*  t3   = (float*) alloc((size_t)NN * 128 * 4);
  __bf16* bxe  = (__bf16*)alloc((size_t)NPAD * 128 * 2);
  __bf16* bxe1 = (__bf16*)alloc((size_t)NPAD * 128 * 2);
  __bf16* brxe = (__bf16*)alloc((size_t)NPAD * 128 * 2);
  __bf16* wb[13];
  const int wsz[13] = {512*128,128*512,128*512,128*128,128*128,128*128,128*128,
                       128*128,128*128,512*128,2*512,512*128,128*512};
  for (int i = 0; i < 13; ++i) wb[i] = (__bf16*)alloc((size_t)wsz[i] * 2);

  auto zero = [&](float* p, size_t n) { k_zero<<<(unsigned)((n + 255) / 256), 256, 0, stream>>>(p, n); };
  auto cvtpad = [&](const float* s, __bf16* d, int C) {
    size_t tot = (size_t)NPAD * C;
    k_cvtpad<<<(unsigned)((tot + 255) / 256), 256, 0, stream>>>(s, d, C);
  };
  auto padzero = [&](__bf16* d, int C) {
    size_t tot = (size_t)(NPAD - NN) * C;
    k_padzero<<<(unsigned)((tot + 255) / 256), 256, 0, stream>>>(d, C);
  };
  auto gemm = [&](const __bf16* A, const __bf16* Wm, const float* b, float* C, int I, int O, int acc) {
    dim3 g(O / 64, NPAD / 128);
    k_gemm_wmma<<<g, 256, 0, stream>>>(A, Wm, b, C, I, O, acc);
  };
  auto stats = [&](const float* x, int C) {
    zero(stat, (size_t)2 * C);
    dim3 g((C + 255) / 256, (NN + 511) / 512);
    k_colstats<<<g, 256, 0, stream>>>(x, stat, C);
  };
  // fused bn + v2v + finalize; dstF / dstB each optional
  auto v2v = [&](const float* x, const float* gg, const float* bt,
                 float* dstF, __bf16* dstB, int C, int relu) {
    zero(vbuf, (size_t)NN * C);
    dim3 g((C + 255) / 256, MM);
    k_v2v<<<g, 256, 0, stream>>>(x, vidx, we, vbuf, stat, gg, bt, C);
    size_t tot = (size_t)NN * C;
    k_vfin<<<(unsigned)((tot + 255) / 256), 256, 0, stream>>>(vbuf, deg, dstF, dstB, C, relu);
    if (dstB) padzero(dstB, C);
  };
  auto ew = [&](size_t n) { return (unsigned)((n + 255) / 256); };

  // weight conversion (small, once per launch)
  for (int i = 0; i < 13; ++i)
    if (i != 10) k_cvt<<<(wsz[i] + 255) / 256, 256, 0, stream>>>(Wp[i], wb[i], (size_t)wsz[i]);

  // degree (shared by every v2v)
  zero(deg, NN);
  k_deg<<<(MM * KK + 255) / 256, 256, 0, stream>>>(vidx, deg, MM * KK);

  // ---- encoder layer 1: x1 = relu(v2v(bn(X @ enc1^T + b)))  -> bf16 directly
  cvtpad(X, abuf, 128);
  gemm(abuf, wb[0], Bp[0], hbuf, 128, 512, 0);
  stats(hbuf, 512);
  v2v(hbuf, G[0], BT[0], nullptr, x1b, 512, 1);

  // ---- encoder layer 2 + fc branch (share bf16(x1))
  gemm(x1b, wb[1], Bp[1], hbuf, 512, 128, 0);    // enc2
  gemm(x1b, wb[2], Bp[2], xe1, 512, 128, 0);     // fc -> xe1 (no BN/v2v)
  stats(hbuf, 128);
  v2v(hbuf, G[1], BT[1], xe, bxe, 128, 0);       // xe (f32 for gating) + bxe (GEMM input)

  // ---- GRU-style gating
  cvtpad(xe1, bxe1, 128);
  gemm(bxe1, wb[3], Bp[3], t1, 128, 128, 0);  gemm(bxe, wb[4], Bp[4], t1, 128, 128, 1);
  gemm(bxe1, wb[5], Bp[5], t2, 128, 128, 0);  gemm(bxe, wb[6], Bp[6], t2, 128, 128, 1);
  k_sigmoid<<<ew((size_t)NN * 128), 256, 0, stream>>>(t1, (size_t)NN * 128);     // z
  k_sigmoid<<<ew((size_t)NN * 128), 256, 0, stream>>>(t2, (size_t)NN * 128);     // r
  k_mul<<<ew((size_t)NN * 128), 256, 0, stream>>>(t3, t2, xe, (size_t)NN * 128); // r*xe
  cvtpad(t3, brxe, 128);
  gemm(bxe1, wb[7], Bp[7], t2, 128, 128, 0);  gemm(brxe, wb[8], Bp[8], t2, 128, 128, 1);
  k_tanh<<<ew((size_t)NN * 128), 256, 0, stream>>>(t2, (size_t)NN * 128);        // o
  k_outs<<<ew((size_t)NN * 128), 256, 0, stream>>>(outs, t1, t2, xe, (size_t)NN * 128);

  // ---- attention-pooled prototype
  stats(outs, 128);
  zero(prot, 128);
  k_proto<<<(NN + 127) / 128, 128, 0, stream>>>(outs, stat, prot);

  // ---- discriminator path
  cvtpad(outs, abuf, 128);
  gemm(abuf, wb[9], Bp[9], hbuf, 128, 512, 0);
  stats(hbuf, 512);
  v2v(hbuf, G[2], BT[2], x1, nullptr, 512, 1);          // f32 h for the O=2 scalar head
  k_gemm_small<<<(NN * 2 + 255) / 256, 256, 0, stream>>>(x1, Wp[10], Bp[10], hbuf, 512);
  stats(hbuf, 2);
  v2v(hbuf, G[3], BT[3], xdis, nullptr, 2, 0);

  // ---- decoder path (abuf still holds bf16(outs) from the dis path)
  gemm(abuf, wb[11], Bp[11], hbuf, 128, 512, 0);
  stats(hbuf, 512);
  v2v(hbuf, G[4], BT[4], nullptr, x1b, 512, 1);
  gemm(x1b, wb[12], Bp[12], hbuf, 512, 128, 0);
  stats(hbuf, 128);
  v2v(hbuf, G[5], BT[5], xde, nullptr, 128, 0);

  (void)in_sizes; (void)n_in; (void)out_size; (void)ws_size;
}
